// LSTMClassifier_89816356094671
// MI455X (gfx1250) — matrix operand order repaired and checked by reading, not compiled, not run on hardware
//
#include <hip/hip_runtime.h>
#include <hip/hip_bf16.h>

#define BATCH 1024
#define SEQ   512
#define HID   64
#define GATES 256   // 4*HID

typedef __bf16 v16bf __attribute__((ext_vector_type(16)));
typedef float  v8f   __attribute__((ext_vector_type(8)));

// ---------------- helpers ----------------

__device__ __forceinline__ unsigned short f32_to_bf16(float f) {
    unsigned int u = __float_as_uint(f);
    unsigned int r = u + 0x7FFFu + ((u >> 16) & 1u);   // round-to-nearest-even
    return (unsigned short)(r >> 16);
}

__device__ __forceinline__ float fast_tanh(float x) {
#if __has_builtin(__builtin_amdgcn_tanhf)
    return __builtin_amdgcn_tanhf(x);                  // native v_tanh_f32
#elif __has_builtin(__builtin_amdgcn_tanh_f32)
    return __builtin_amdgcn_tanh_f32(x);
#else
    float t = __builtin_amdgcn_exp2f(x * 2.885390081777927f);
    return (t - 1.0f) * __builtin_amdgcn_rcpf(t + 1.0f);
#endif
}

// Build a v16bf A/B fragment from two 16-byte segments.
__device__ __forceinline__ v16bf load_frag(const unsigned short* p0,
                                           const unsigned short* p1) {
    union { v16bf v; uint4 q[2]; } f;
    f.q[0] = *(const uint4*)p0;
    f.q[1] = *(const uint4*)p1;
    return f.v;
}

// A-matrix 16x32 bf16 (row-major source, row stride rs elements).
// lane<16: row=lane, K = {kc*32+0..7, kc*32+16..23}; lane>=16: K shifted by +8.
__device__ __forceinline__ v16bf load_a_frag(const unsigned short* base,
                                             int rs, int kc, int lane) {
    const int m    = lane & 15;
    const int half = (lane >> 4) * 8;
    const unsigned short* p = base + (size_t)m * rs + kc * 32 + half;
    return load_frag(p, p + 16);
}

// B-matrix 32x16 bf16 from weight matrix W (GATES x 64, row-major):
// B[k][n] = W[g0+n][kbase+k]; lane<16: K = {0..7, 16..23}; lane>=16: K shifted by +8.
__device__ __forceinline__ v16bf load_b_frag(const unsigned short* w,
                                             int g0, int kc, int lane) {
    const int n     = lane & 15;
    const int khalf = (lane >> 4) * 8;
    const unsigned short* p = w + (size_t)(g0 + n) * HID + kc * 32 + khalf;
    return load_frag(p, p + 16);
}

__device__ __forceinline__ v8f wmma_bf16(v16bf a, v16bf b, v8f c) {
    return __builtin_amdgcn_wmma_f32_16x16x32_bf16(
        false, a, false, b, (short)0, c, false, false);
}

// ---------------- kernels ----------------

__global__ __launch_bounds__(256) void cvt_bf16_kernel(
    const float* __restrict__ in, unsigned short* __restrict__ out, int n) {
    for (int i = blockIdx.x * blockDim.x + threadIdx.x; i < n;
         i += gridDim.x * blockDim.x)
        out[i] = f32_to_bf16(in[i]);
}

__global__ __launch_bounds__(256) void prep_weights_kernel(
    const float* __restrict__ w_ih0, const float* __restrict__ w_hh0,
    const float* __restrict__ b_ih0, const float* __restrict__ b_hh0,
    const float* __restrict__ w_ih1, const float* __restrict__ w_hh1,
    const float* __restrict__ b_ih1, const float* __restrict__ b_hh1,
    unsigned short* __restrict__ wih0b, unsigned short* __restrict__ whh0b,
    unsigned short* __restrict__ wih1b, unsigned short* __restrict__ whh1b,
    float* __restrict__ bias0, float* __restrict__ bias1) {
    const int i = blockIdx.x * blockDim.x + threadIdx.x;
    if (i < GATES * HID) {
        wih0b[i] = f32_to_bf16(w_ih0[i]);
        whh0b[i] = f32_to_bf16(w_hh0[i]);
        wih1b[i] = f32_to_bf16(w_ih1[i]);
        whh1b[i] = f32_to_bf16(w_hh1[i]);
    }
    if (i < GATES) {
        bias0[i] = b_ih0[i] + b_hh0[i];
        bias1[i] = b_ih1[i] + b_hh1[i];
    }
}

// Fused LSTM layer: block owns a 16-row batch tile for all `seq` steps.
// Wave w owns gate columns [32w, 32w+32): 8 v_wmma per wave per step,
// x-fragments software-pipelined one step ahead. Activations are uniform:
//   sigmoid(x) = 0.5*tanh(0.5x)+0.5, tanh(x) = tanh(x)
// -> v = fma(tanh(x*sa), sb, sc) with wave-uniform (sa,sb,sc); no branches.
__global__ __launch_bounds__(256) void lstm_scan_kernel(
    const unsigned short* __restrict__ x,     // (B, seq, 64) bf16
    const unsigned short* __restrict__ w_ih,  // (256, 64) bf16
    const unsigned short* __restrict__ w_hh,  // (256, 64) bf16
    const float* __restrict__ bias,           // (256) f32, b_ih+b_hh
    unsigned short* __restrict__ h_seq,       // (B, seq, 64) bf16 (if write_seq)
    float* __restrict__ h_final,              // (B, 64) f32, t = seq-1
    int seq, int write_seq) {
    __shared__ float gates[16][GATES + 4];    // stride 260 dwords: no bank conflicts
    __shared__ unsigned short hbuf[16][HID + 8];

    const int tid  = threadIdx.x;
    const int lane = tid & 31;
    const int wave = tid >> 5;
    const int b0   = blockIdx.x * 16;

    for (int i = tid; i < 16 * (HID + 8); i += 256)
        (&hbuf[0][0])[i] = 0;

    // Resident B fragments for this wave's 32 gate columns.
    const int g0 = wave * 32;
    v16bf bih[2][2], bhh[2][2];
#pragma unroll
    for (int nt = 0; nt < 2; ++nt)
#pragma unroll
        for (int kc = 0; kc < 2; ++kc) {
            bih[nt][kc] = load_b_frag(w_ih, g0 + nt * 16, kc, lane);
            bhh[nt][kc] = load_b_frag(w_hh, g0 + nt * 16, kc, lane);
        }
    float bias_v[2];
    bias_v[0] = bias[g0 + (lane & 15)];
    bias_v[1] = bias[g0 + 16 + (lane & 15)];

    // Wave-uniform activation constants: g-gate (waves 4,5) -> plain tanh,
    // i/f/o gates -> sigmoid via tanh identity.
    const int is_ggate = ((wave >> 1) == 2);
    const float sa = is_ggate ? 1.0f : 0.5f;
    const float sb = is_ggate ? 1.0f : 0.5f;
    const float sc = is_ggate ? 0.0f : 0.5f;

    // Cell state: 4 consecutive (row, hid) elements per thread.
    float cst[4] = {0.f, 0.f, 0.f, 0.f};
    const int frow = (tid * 4) >> 6;   // 0..15
    const int fcol = (tid * 4) & 63;   // 0,4,...,60

    const unsigned short* xrow = x + (size_t)b0 * seq * HID;
    const int rbase = (lane >> 4) * 8;
    const int xrs   = seq * HID;

    // Prime the x-fragment pipeline (t = 0).
    v16bf ax0 = load_a_frag(xrow, xrs, 0, lane);
    v16bf ax1 = load_a_frag(xrow, xrs, 1, lane);

    __syncthreads();

    for (int t = 0; t < seq; ++t) {
        // h_{t-1} fragments from LDS.
        v16bf ah0 = load_a_frag(&hbuf[0][0], HID + 8, 0, lane);
        v16bf ah1 = load_a_frag(&hbuf[0][0], HID + 8, 1, lane);

        v8f acc[2];
#pragma unroll
        for (int nt = 0; nt < 2; ++nt) {
#pragma unroll
            for (int j = 0; j < 8; ++j) acc[nt][j] = bias_v[nt];
            acc[nt] = wmma_bf16(ax0, bih[nt][0], acc[nt]);
            acc[nt] = wmma_bf16(ax1, bih[nt][1], acc[nt]);
            acc[nt] = wmma_bf16(ah0, bhh[nt][0], acc[nt]);
            acc[nt] = wmma_bf16(ah1, bhh[nt][1], acc[nt]);
        }

        // Prefetch next step's x fragments (overlaps with activations+update).
        if (t + 1 < seq) {
            const unsigned short* xt = xrow + (size_t)(t + 1) * HID;
            ax0 = load_a_frag(xt, xrs, 0, lane);
            ax1 = load_a_frag(xt, xrs, 1, lane);
        }

        // Uniform branch-free gate nonlinearity: fma(tanh(x*sa), sb, sc).
#pragma unroll
        for (int nt = 0; nt < 2; ++nt) {
            const int col = g0 + nt * 16 + (lane & 15);
#pragma unroll
            for (int j = 0; j < 8; ++j) {
                gates[rbase + j][col] = fmaf(fast_tanh(acc[nt][j] * sa), sb, sc);
            }
        }
        __syncthreads();

        // Elementwise c/h update: 4 hidden elements of row frow, vectorized.
        {
            const float4 iv = *(const float4*)&gates[frow][fcol];
            const float4 fv = *(const float4*)&gates[frow][64 + fcol];
            const float4 gv = *(const float4*)&gates[frow][128 + fcol];
            const float4 ov = *(const float4*)&gates[frow][192 + fcol];
            float hv[4];
            cst[0] = fv.x * cst[0] + iv.x * gv.x;
            cst[1] = fv.y * cst[1] + iv.y * gv.y;
            cst[2] = fv.z * cst[2] + iv.z * gv.z;
            cst[3] = fv.w * cst[3] + iv.w * gv.w;
            hv[0] = ov.x * fast_tanh(cst[0]);
            hv[1] = ov.y * fast_tanh(cst[1]);
            hv[2] = ov.z * fast_tanh(cst[2]);
            hv[3] = ov.w * fast_tanh(cst[3]);

            uint2 pk;
            pk.x = (unsigned)f32_to_bf16(hv[0]) | ((unsigned)f32_to_bf16(hv[1]) << 16);
            pk.y = (unsigned)f32_to_bf16(hv[2]) | ((unsigned)f32_to_bf16(hv[3]) << 16);
            *(uint2*)&hbuf[frow][fcol] = pk;
            if (write_seq)
                *(uint2*)&h_seq[((size_t)(b0 + frow) * seq + t) * HID + fcol] = pk;
            if (t == seq - 1) {
                float4 hf4 = make_float4(hv[0], hv[1], hv[2], hv[3]);
                *(float4*)&h_final[(size_t)(b0 + frow) * HID + fcol] = hf4;
            }
        }
        __syncthreads();
    }
}

// out[b] = dot(h_final[b,:], fc_w) + fc_b
__global__ __launch_bounds__(256) void fc_kernel(
    const float* __restrict__ hf, const float* __restrict__ fc_w,
    const float* __restrict__ fc_b, float* __restrict__ out, int n) {
    const int b = blockIdx.x * blockDim.x + threadIdx.x;
    if (b < n) {
        float s = fc_b[0];
#pragma unroll
        for (int k = 0; k < HID; ++k) s += hf[(size_t)b * HID + k] * fc_w[k];
        out[b] = s;
    }
}

// ---------------- launcher ----------------

extern "C" void kernel_launch(void* const* d_in, const int* in_sizes, int n_in,
                              void* d_out, int out_size, void* d_ws, size_t ws_size,
                              hipStream_t stream) {
    const float* x     = (const float*)d_in[0];
    const float* w_ih0 = (const float*)d_in[1];
    const float* w_hh0 = (const float*)d_in[2];
    const float* b_ih0 = (const float*)d_in[3];
    const float* b_hh0 = (const float*)d_in[4];
    const float* w_ih1 = (const float*)d_in[5];
    const float* w_hh1 = (const float*)d_in[6];
    const float* b_ih1 = (const float*)d_in[7];
    const float* b_hh1 = (const float*)d_in[8];
    const float* fc_w  = (const float*)d_in[9];
    const float* fc_b  = (const float*)d_in[10];
    float* out = (float*)d_out;

    char* p = (char*)d_ws;
    auto carve = [&](size_t bytes) {
        char* r = p;
        p += (bytes + 255) & ~(size_t)255;
        return r;
    };
    const size_t nx = (size_t)BATCH * SEQ * HID;
    unsigned short* x_bf  = (unsigned short*)carve(nx * 2);
    unsigned short* h1_bf = (unsigned short*)carve(nx * 2);
    unsigned short* wih0b = (unsigned short*)carve(GATES * HID * 2);
    unsigned short* whh0b = (unsigned short*)carve(GATES * HID * 2);
    unsigned short* wih1b = (unsigned short*)carve(GATES * HID * 2);
    unsigned short* whh1b = (unsigned short*)carve(GATES * HID * 2);
    float* bias0 = (float*)carve(GATES * 4);
    float* bias1 = (float*)carve(GATES * 4);
    float* hfin0 = (float*)carve((size_t)BATCH * HID * 4);
    float* hfin1 = (float*)carve((size_t)BATCH * HID * 4);

    cvt_bf16_kernel<<<8192, 256, 0, stream>>>(x, x_bf, (int)nx);

    prep_weights_kernel<<<(GATES * HID + 255) / 256, 256, 0, stream>>>(
        w_ih0, w_hh0, b_ih0, b_hh0, w_ih1, w_hh1, b_ih1, b_hh1,
        wih0b, whh0b, wih1b, whh1b, bias0, bias1);

    lstm_scan_kernel<<<BATCH / 16, 256, 0, stream>>>(
        x_bf, wih0b, whh0b, bias0, h1_bf, hfin0, SEQ, 1);

    lstm_scan_kernel<<<BATCH / 16, 256, 0, stream>>>(
        h1_bf, wih1b, whh1b, bias1, h1_bf /*unused*/, hfin1, SEQ, 0);

    fc_kernel<<<(BATCH + 255) / 256, 256, 0, stream>>>(hfin1, fc_w, fc_b, out, BATCH);
}
